// DeconvCapsuleLayer_90443421319794
// MI455X (gfx1250) — matrix operand order repaired and checked by reading, not compiled, not run on hardware
//
#include <hip/hip_runtime.h>

// ---------------------------------------------------------------------------
// Fused SegCaps deconv-capsule layer for gfx1250 (MI455X).
//  Stage 1: gather stride-2 transposed-conv input patches (per parity class)
//           into LDS as f16  (A tile: 64 rows x 128 K)
//  Stage 2: gather weight fragments in WMMA B layout (f32 -> f16)
//  Stage 3: GEMM via v_wmma_f32_16x16x32_f16 (f32 accumulate) -> votes in LDS
//  Stage 4: 3-round dynamic routing (softmax/squash) per pixel, wave-parallel
// Votes never touch HBM: traffic = x + W + act ~ 40 MB  => memory-roofline.
// ---------------------------------------------------------------------------

typedef _Float16 v16h __attribute__((ext_vector_type(16)));
typedef _Float16 h8   __attribute__((ext_vector_type(8)));
typedef _Float16 h4   __attribute__((ext_vector_type(4)));
typedef float    v8f  __attribute__((ext_vector_type(8)));

#define BATCH   4
#define HIN     56
#define HOUT    112
#define CIN     8
#define AIN     32
#define COUT    8
#define AOUT    16
#define NCH     128           // COUT*AOUT
#define KDIM    128           // 4 taps * 32 in-atoms
#define GRP_PER_CLASS (BATCH*HIN*HIN)   // 12544 pixel groups per parity class
#define GRP_PER_BLK   8                  // 8 groups (64 M-rows) per workgroup
#define BLK_PER_CLASS (GRP_PER_CLASS/GRP_PER_BLK)  // 1568

#define A_STRIDE 136          // 128 + 8 halves pad  -> conflict-free b128 reads
#define V_STRIDE 132          // 128 + 4 floats pad  -> conflict-free acc writes

__global__ __launch_bounds__(256)
void segcaps_deconv_route(const float* __restrict__ x,   // [4,56,56,8,32]
                          const float* __restrict__ Wt,  // [4,4,128,32]
                          const float* __restrict__ bias,// [8,16]
                          float* __restrict__ out)       // [4,112,112,8,16]
{
    __shared__ _Float16 lA[64 * A_STRIDE];   // 17.0 KB
    __shared__ float    lV[64 * V_STRIDE];   // 33.8 KB

    const int tid  = threadIdx.x;
    const int lane = tid & 31;
    const int wave = tid >> 5;

    const int p    = blockIdx.x / BLK_PER_CLASS;   // parity class
    const int gblk = blockIdx.x % BLK_PER_CLASS;
    const int ph = p >> 1, pw = p & 1;
    const int g0 = gblk * GRP_PER_BLK;

    // ---------------- Stage 1: A tile -> LDS (f32 load, cvt f16) ------------
    {
        const int r   = tid >> 2;        // 0..63  (M row)
        const int tap = tid & 3;         // 0..3
        const int th = tap >> 1, tw = tap & 1;
        const int gl = r >> 3, ip = r & 7;
        const int g  = g0 + gl;
        const int bp  = g / (HIN * HIN);
        const int rem = g % (HIN * HIN);
        const int hop = rem / HIN, wop = rem % HIN;
        const int n  = bp * 8 + ip;          // conv batch (scrambled reshape)
        const int xb = n & 3, xc = n >> 2;   // x[b], x[cin]
        int hi = ph ? (hop + th) : (hop - th);
        int wi = pw ? (wop + tw) : (wop - tw);
        const bool ok = ((unsigned)hi < (unsigned)HIN) && ((unsigned)wi < (unsigned)HIN);
        const float msk = ok ? 1.0f : 0.0f;
        if (!ok) { hi = 0; wi = 0; }         // keep address in-bounds
        const float4* src = (const float4*)(x +
            ((((size_t)xb * HIN + hi) * HIN + wi) * CIN + xc) * AIN);
        _Float16* dst = &lA[r * A_STRIDE + tap * 32];
        #pragma unroll
        for (int q = 0; q < 8; ++q) {
            float4 v = src[q];
            h4 h = { (_Float16)(v.x * msk), (_Float16)(v.y * msk),
                     (_Float16)(v.z * msk), (_Float16)(v.w * msk) };
            *(h4*)(dst + q * 4) = h;
        }
    }

    // ---------------- Stage 2: B fragments (WMMA B layout) ------------------
    // B elem j of v16h: K = kk*32 + (lane<16 ? j : 16+j), N = lane%16.
    // K = tap*32 + ic  =>  tap == kk, ic = (lane<16?0:16)+j  -> 64B contiguous.
    v16h bfrag[4];
    {
        const int oc  = wave * 16 + (lane & 15);
        const int icb = (lane < 16) ? 0 : 16;
        #pragma unroll
        for (int kk = 0; kk < 4; ++kk) {
            const int th = kk >> 1, tw = kk & 1;
            const int kh = ph ? (th ? 0 : 2) : (th ? 3 : 1);
            const int kw = pw ? (tw ? 0 : 2) : (tw ? 3 : 1);
            const float4* wp4 = (const float4*)(Wt +
                (((size_t)(kh * 4 + kw) * NCH + oc) * AIN + icb));
            v16h bf;
            #pragma unroll
            for (int q = 0; q < 4; ++q) {
                float4 v = wp4[q];
                bf[4*q+0] = (_Float16)v.x;  bf[4*q+1] = (_Float16)v.y;
                bf[4*q+2] = (_Float16)v.z;  bf[4*q+3] = (_Float16)v.w;
            }
            bfrag[kk] = bf;
        }
    }

    __syncthreads();

    // ---------------- Stage 3: WMMA GEMM -> votes in LDS --------------------
    {
        const int hhalf = (lane < 16) ? 0 : 8;
        #pragma unroll
        for (int m = 0; m < 4; ++m) {
            v8f acc = {0.f,0.f,0.f,0.f,0.f,0.f,0.f,0.f};
            const int row = m * 16 + (lane & 15);
            const int ab  = row * A_STRIDE + hhalf * 2;
            #pragma unroll
            for (int kk = 0; kk < 4; ++kk) {
                h8 lo  = *(const h8*)&lA[ab + kk * 32];
                h8 hi8 = *(const h8*)&lA[ab + kk * 32 + 8];
                v16h af = __builtin_shufflevector(lo, hi8,
                    0,1,2,3,4,5,6,7,8,9,10,11,12,13,14,15);
                acc = __builtin_amdgcn_wmma_f32_16x16x32_f16(
                    false, af, false, bfrag[kk], (short)0, acc, false, false);
            }
            #pragma unroll
            for (int v = 0; v < 8; ++v)
                lV[(m * 16 + v + hhalf) * V_STRIDE + wave * 16 + (lane & 15)] = acc[v];
        }
    }

    __syncthreads();

    // ---------------- Stage 4: dynamic routing (wave per pixel) -------------
    // lane owns cols {lane + 32q}: c_q = 2q + (lane>=16), a = lane&15.
    {
        float vv[8][4];
        #pragma unroll
        for (int i = 0; i < 8; ++i)
            #pragma unroll
            for (int q = 0; q < 4; ++q)
                vv[i][q] = lV[(wave * 8 + i) * V_STRIDE + lane + 32 * q];

        const int a = lane & 15;
        const int chi = (lane >= 16) ? 1 : 0;
        float bq[4];
        #pragma unroll
        for (int q = 0; q < 4; ++q)
            bq[q] = bias[(2 * q + chi) * AOUT + a];

        float logit[8][4];
        #pragma unroll
        for (int i = 0; i < 8; ++i)
            #pragma unroll
            for (int q = 0; q < 4; ++q) logit[i][q] = 0.f;

        float act[4];
        #pragma unroll
        for (int it = 0; it < 3; ++it) {
            float route[8][4];
            if (it == 0) {
                #pragma unroll
                for (int i = 0; i < 8; ++i)
                    #pragma unroll
                    for (int q = 0; q < 4; ++q) route[i][q] = 0.125f;
            } else {
                #pragma unroll
                for (int i = 0; i < 8; ++i) {
                    float oth[4], e[4];
                    #pragma unroll
                    for (int q = 0; q < 4; ++q) oth[q] = __shfl_xor(logit[i][q], 16);
                    float m = -3.0e38f;
                    #pragma unroll
                    for (int q = 0; q < 4; ++q)
                        m = fmaxf(m, fmaxf(logit[i][q], oth[q]));
                    float den = 0.f;
                    #pragma unroll
                    for (int q = 0; q < 4; ++q) {
                        e[q] = __expf(logit[i][q] - m);
                        den += e[q] + __expf(oth[q] - m);
                    }
                    #pragma unroll
                    for (int q = 0; q < 4; ++q) route[i][q] = e[q] / den;
                }
            }
            #pragma unroll
            for (int q = 0; q < 4; ++q) {
                float pre = bq[q];
                #pragma unroll
                for (int i = 0; i < 8; ++i) pre += route[i][q] * vv[i][q];
                float sq = pre * pre;                 // reduce over 16 atoms
                sq += __shfl_xor(sq, 1);
                sq += __shfl_xor(sq, 2);
                sq += __shfl_xor(sq, 4);
                sq += __shfl_xor(sq, 8);
                float scale = sq / ((1.0f + sq) * sqrtf(sq + 1e-9f));
                act[q] = pre * scale;
            }
            if (it < 2) {
                #pragma unroll
                for (int i = 0; i < 8; ++i)
                    #pragma unroll
                    for (int q = 0; q < 4; ++q) {
                        float d = vv[i][q] * act[q];
                        d += __shfl_xor(d, 1);
                        d += __shfl_xor(d, 2);
                        d += __shfl_xor(d, 4);
                        d += __shfl_xor(d, 8);
                        logit[i][q] += d;
                    }
            }
        }

        const int g   = g0 + wave;
        const int bp  = g / (HIN * HIN);
        const int rem = g % (HIN * HIN);
        const int ho  = 2 * (rem / HIN) + ph;
        const int wo  = 2 * (rem % HIN) + pw;
        float* ob = out + (((size_t)bp * HOUT + ho) * HOUT + wo) * NCH;
        #pragma unroll
        for (int q = 0; q < 4; ++q)
            ob[(2 * q + chi) * AOUT + a] = act[q];
    }
}

extern "C" void kernel_launch(void* const* d_in, const int* in_sizes, int n_in,
                              void* d_out, int out_size, void* d_ws, size_t ws_size,
                              hipStream_t stream) {
    (void)in_sizes; (void)n_in; (void)d_ws; (void)ws_size; (void)out_size;
    const float* x  = (const float*)d_in[0];
    const float* Wt = (const float*)d_in[1];
    const float* b  = (const float*)d_in[2];
    float* out = (float*)d_out;
    dim3 grid(4 * BLK_PER_CLASS);   // 6272 workgroups
    dim3 block(256);                // 8 wave32 waves
    segcaps_deconv_route<<<grid, block, 0, stream>>>(x, Wt, b, out);
}
